// NgramReranker_51281909514707
// MI455X (gfx1250) — compile-verified, matrix operand order checked by reading, not run on hardware
//
#include <hip/hip_runtime.h>
#include <hip/hip_bf16.h>

typedef __attribute__((ext_vector_type(16))) _Float16 v16h;
typedef __attribute__((ext_vector_type(8)))  float    v8f;

#define B_  64
#define C_  16
#define LC_ 64
#define LS_ 48

// Whole-wave pairwise masked-equality count:
//   returns  sum_{i<Mlen, j<Nlen} [ R[i]==Ccol[j] && R[i]!=pad && Ccol[j]!=pad ]
// f16 0/1 indicator tiles in the documented 16x32 A layout, reduced with
// v_wmma_f32_16x16x32_f16 against an all-ones B (B-layout independent);
// the C accumulator chains across all row-tiles and col-blocks.
//
// Mask folding: invalid/pad rows use sentinel rtok = -1 (never equals any
// token >= 0); out-of-range columns use sentinel ctok = pad (never equals a
// valid non-pad row token). So each element is ONE compare + ONE cndmask.
__device__ __forceinline__ float wave_pairwise_eq_sum(
    const int* __restrict__ R, int Mlen,
    const int* __restrict__ Ccol, int Nlen, int pad)
{
  const int lane   = threadIdx.x & 31;
  const int m      = lane & 15;
  const int laneHi = lane >> 4;   // 0: lanes 0-15, 1: lanes 16-31

  v16h onesB;
#pragma unroll
  for (int i = 0; i < 16; ++i) onesB[i] = (_Float16)1.0f;

  v8f acc = {0.f, 0.f, 0.f, 0.f, 0.f, 0.f, 0.f, 0.f};

  for (int rb = 0; rb < Mlen; rb += 16) {
    const int ridx = rb + m;
    int rtok = (ridx < Mlen) ? R[ridx] : -1;
    if (rtok == pad) rtok = -1;            // fold both row masks into the token
    for (int cb = 0; cb < Nlen; cb += 32) {
      v16h a;
#pragma unroll
      for (int v = 0; v < 8; ++v) {
#pragma unroll
        for (int h = 0; h < 2; ++h) {
          // 16-bit A 16x32 layout: K = 2v + h (+8 for lanes>=16) (+8 for v>=4)
          const int K    = 2*v + h + (laneHi ? 8 : 0) + ((v >= 4) ? 8 : 0);
          const int cidx = cb + K;
          const int ctok = (cidx < Nlen) ? Ccol[cidx] : pad;  // sentinel never matches
          a[2*v + h] = (ctok == rtok) ? (_Float16)1.0f : (_Float16)0.0f;
        }
      }
      acc = __builtin_amdgcn_wmma_f32_16x16x32_f16(
          /*neg_a=*/false, a, /*neg_b=*/false, onesB,
          /*c_mod=*/(short)0, acc, /*reuse_a=*/false, /*reuse_b=*/false);
    }
  }
  // D[m,n] identical for all n (ones B). Lane holds rows m = r + 8*laneHi.
  float s = 0.f;
#pragma unroll
  for (int r = 0; r < 8; ++r) s += acc[r];
  s += __shfl_xor(s, 16, 32);   // add the other 8-row half -> total in every lane
  return s;
}

__global__ __launch_bounds__(512)
void ngram_rerank_kernel(const int* __restrict__ candidates,
                         const int* __restrict__ src_tokens,
                         const int* __restrict__ pad_ptr,
                         float* __restrict__ out)
{
  __shared__ int   sSrc[LS_];
  __shared__ int   sCand[C_ * LC_];
  __shared__ float sDots[C_];
  __shared__ float sCandN[C_];
  __shared__ float sRefN;
  __shared__ float sScores[C_];
  __shared__ int   sOrder[C_];

  const int b   = blockIdx.x;
  const int tid = threadIdx.x;
  const int pad = pad_ptr[0];

  // Stage this batch row's tokens in LDS (values < 32000).
  for (int i = tid; i < C_ * LC_; i += (int)blockDim.x)
    sCand[i] = candidates[(size_t)b * (C_ * LC_) + i];
  for (int i = tid; i < LS_; i += (int)blockDim.x)
    sSrc[i] = src_tokens[(size_t)b * LS_ + i];
  __syncthreads();

  // 16 waves: wave c handles candidate c (dots + cand norm^2); wave 0 also ref norm^2.
  const int wave = tid >> 5;
  if (wave < C_) {
    const int c = wave;
    const float d  = wave_pairwise_eq_sum(sSrc, LS_, sCand + c * LC_, LC_, pad);
    const float cn = wave_pairwise_eq_sum(sCand + c * LC_, LC_, sCand + c * LC_, LC_, pad);
    if ((tid & 31) == 0) { sDots[c] = d; sCandN[c] = cn; }
  }
  if (wave == 0) {
    const float rn = wave_pairwise_eq_sum(sSrc, LS_, sSrc, LS_, pad);
    if ((tid & 31) == 0) sRefN = rn;
  }
  __syncthreads();

  // scores = 1 - dots / (||ref|| * ||cand||)
  if (tid < C_) {
    const float denom = sqrtf(sRefN) * sqrtf(sCandN[tid]);
    sScores[tid] = 1.0f - sDots[tid] / denom;
  }
  __syncthreads();

  // Stable descending sort of 16 scores (ties -> lower index first),
  // matching argsort(-scores) semantics.
  if (tid == 0) {
    int ord[C_];
    for (int i = 0; i < C_; ++i) ord[i] = i;
    for (int i = 1; i < C_; ++i) {
      const int   oi = ord[i];
      const float si = sScores[oi];
      int j = i - 1;
      while (j >= 0 && sScores[ord[j]] < si) { ord[j + 1] = ord[j]; --j; }
      ord[j + 1] = oi;
    }
    for (int i = 0; i < C_; ++i) sOrder[i] = ord[i];
  }
  __syncthreads();

  const int best = sOrder[0];

  // Output layout (flat, return order): output[B,LC] | out_lengths[B] | sorted_scores[B,C]
  for (int j = tid; j < LC_; j += (int)blockDim.x)
    out[(size_t)b * LC_ + j] = (float)sCand[best * LC_ + j];

  if (tid == 0) {
    int len = 0;
    for (int j = 0; j < LC_; ++j) len += (sCand[best * LC_ + j] != pad) ? 1 : 0;
    out[(size_t)B_ * LC_ + b] = (float)len;
  }

  if (tid < C_)
    out[(size_t)B_ * LC_ + B_ + (size_t)b * C_ + tid] = sScores[sOrder[tid]];
}

extern "C" void kernel_launch(void* const* d_in, const int* in_sizes, int n_in,
                              void* d_out, int out_size, void* d_ws, size_t ws_size,
                              hipStream_t stream) {
  (void)in_sizes; (void)n_in; (void)out_size; (void)d_ws; (void)ws_size;
  const int* cand = (const int*)d_in[0];   // candidates [64,16,64] int
  const int* src  = (const int*)d_in[1];   // src_tokens [64,48] int
  const int* padp = (const int*)d_in[3];   // pad_id scalar
  float* out = (float*)d_out;
  ngram_rerank_kernel<<<B_, 512, 0, stream>>>(cand, src, padp, out);
}
